// Irregular_process_lenth_88665304858970
// MI455X (gfx1250) — matrix operand order repaired and checked by reading, not compiled, not run on hardware
//
#include <hip/hip_runtime.h>
#include <hip/hip_bf16.h>

// ---- sizes from the reference ----
#define BB   32
#define CC   64
#define LL   512
#define HID  128

typedef __attribute__((ext_vector_type(16))) _Float16 v16h;
typedef __attribute__((ext_vector_type(8)))  _Float16 v8h;
typedef __attribute__((ext_vector_type(8)))  float    v8f;

// --------------------------------------------------------------------------
// Kernel A: W3 (fp32, [k=128][f=65536] row-major) -> W3t (f16, [f][k])
// Tiled LDS transpose: 64 f x 32 k per block, 256 threads.
// --------------------------------------------------------------------------
__global__ void __launch_bounds__(256) w3_convert_kernel(const float* __restrict__ W3,
                                                         _Float16* __restrict__ W3t) {
  __shared__ float tile[32][64 + 1];
  const int fi = blockIdx.x * 64;   // f tile base (0..65472)
  const int ki = blockIdx.y * 32;   // k tile base (0..96)
  const int t  = threadIdx.x;

  // coalesced read: consecutive f within a row of W3
  {
    const int fl  = t & 63;
    const int kl0 = t >> 6;         // 0..3
#pragma unroll
    for (int i = 0; i < 8; ++i) {
      const int kk = kl0 + i * 4;   // 0..31
      tile[kk][fl] = W3[(size_t)(ki + kk) * (LL * HID) + fi + fl];
    }
  }
  __syncthreads();
  // coalesced write: consecutive k within a row of W3t
  {
    const int kl  = t & 31;
    const int fl0 = t >> 5;         // 0..7
#pragma unroll
    for (int i = 0; i < 8; ++i) {
      const int ff = fl0 + i * 8;   // 0..63
      W3t[(size_t)(fi + ff) * HID + ki + kl] = (_Float16)tile[kl][ff];
    }
  }
}

// --------------------------------------------------------------------------
// Kernel B: h2 = relu(relu(X@W1+b1)@W2+b2), stored f16 row-major [bc][k=128].
// One block per (b,c) row, 128 threads (one per output feature).
// --------------------------------------------------------------------------
__global__ void __launch_bounds__(128) mlp_kernel(const float* __restrict__ X,
                                                  const float* __restrict__ W1,
                                                  const float* __restrict__ b1,
                                                  const float* __restrict__ W2,
                                                  const float* __restrict__ b2,
                                                  _Float16* __restrict__ H2) {
  __shared__ float xs[LL];
  __shared__ float h1s[HID];
  const int bc = blockIdx.x;      // 0..2047
  const int j  = threadIdx.x;     // 0..127
  const float* xrow = X + (size_t)bc * LL;
#pragma unroll
  for (int i = 0; i < 4; ++i) xs[j + i * 128] = xrow[j + i * 128];
  __syncthreads();
  float a = b1[j];
  for (int l = 0; l < LL; ++l) a = fmaf(xs[l], W1[l * HID + j], a);
  h1s[j] = fmaxf(a, 0.f);
  __syncthreads();
  float a2 = b2[j];
  for (int k = 0; k < HID; ++k) a2 = fmaf(h1s[k], W2[k * HID + j], a2);
  H2[(size_t)bc * HID + j] = (_Float16)fmaxf(a2, 0.f);
}

// --------------------------------------------------------------------------
// Kernel C (main): fused filt-GEMM + channel softmax + weighted L-reduction.
// Block = one (b, h) pair; 256 threads = 8 waves.
//   Z[c][l] = h2[b,c,:] . W3[:, h*L+l]        (b3 cancels in softmax over c)
//   P[:,l]  = softmax_c Z[:,l]
//   Xout[b,c,h] = sum_l X[b,c,l] * P[c,l]
// Waves tile Z as 4(M=c) x 2(N=l) 16x16 tiles, K=128 in 4 WMMA steps,
// L processed in 16 chunks of 32, B-fragments double-buffered so loads of
// chunk n+1 overlap WMMA + softmax of chunk n. X chunk is staged to LDS with
// GLOBAL_LOAD_ASYNC_TO_LDS_B128 (ASYNCcnt path).
// --------------------------------------------------------------------------
__global__ void __launch_bounds__(256) filt_apply_kernel(const float* __restrict__ X,
                                                         const _Float16* __restrict__ H2,
                                                         const _Float16* __restrict__ W3t,
                                                         float* __restrict__ Xout) {
  const int h = blockIdx.x;   // 0..127  (hid)
  const int b = blockIdx.y;   // 0..31
  const int t = threadIdx.x;  // 0..255
  const int wave = t >> 5, lane = t & 31;
  const int mi = wave >> 1, ni = wave & 1;   // M-tile 0..3, N-tile 0..1
  const int lhalf = lane >> 4;               // 0/1
  const int lmod  = lane & 15;

  __shared__ float Zs[CC * 32];
  __shared__ float Ps[CC * 32];
  __shared__ float Xs[CC * 32];
  __shared__ float pred[8 * 32];
  __shared__ float colmax[32];
  __shared__ float colsum[32];
  __shared__ float racc[CC * 4];

  // ---- preload A fragments (contiguous k = 16*lhalf + s, matching B), reused all chunks
  v16h afrag[4];
  {
    const int row = mi * 16 + lmod;                       // c row
    const _Float16* arow = H2 + ((size_t)(b * CC + row)) * HID;
    const int khb = lhalf * 16;
#pragma unroll
    for (int ks = 0; ks < 4; ++ks) {
      const _Float16* p0 = arow + ks * 32 + khb;
      v8h lo = *(const v8h*)(p0);
      v8h hi = *(const v8h*)(p0 + 8);
      afrag[ks] = __builtin_shufflevector(lo, hi, 0,1,2,3,4,5,6,7,8,9,10,11,12,13,14,15);
    }
  }

  // ---- B fragment loader (contiguous k = 16*lhalf + s, same order as A): 8x b128 per call
  const _Float16* bbase = W3t + ((size_t)h * LL + ni * 16 + lmod) * HID + lhalf * 16;
  auto load_b = [&](int chunk, v16h* dst) {
    const _Float16* bcol = bbase + (size_t)chunk * 32 * HID;
#pragma unroll
    for (int ks = 0; ks < 4; ++ks) dst[ks] = *(const v16h*)(bcol + ks * 32);
  };

  const float* xb = X + (size_t)b * CC * LL;
  float acc_out = 0.f;
  const int q    = t >> 6;   // 0..3  (l quarter for accumulation)
  const int cacc = t & 63;   // c     (row for accumulation)

  // async X staging addresses (thread t owns 8 consecutive floats of Xs)
  const int cx = t >> 2;
  const int l0 = (t & 3) * 8;
  const unsigned ldsXs = (unsigned)(uintptr_t)(&Xs[cx * 32 + l0]);
  const float* gx = xb + (size_t)cx * LL + l0;

  v16h bcur[4], bnext[4];
  load_b(0, bcur);

  for (int chunk = 0; chunk < 16; ++chunk) {
    // ---- prefetch next chunk's B fragments (overlaps WMMA + softmax below)
    if (chunk < 15) load_b(chunk + 1, bnext);
    // ---- stage X chunk directly into LDS via async copy (ASYNCcnt)
    {
      const float* gp = gx + chunk * 32;
      asm volatile(
          "global_load_async_to_lds_b128 %0, %2, off\n\t"
          "global_load_async_to_lds_b128 %1, %3, off"
          :
          : "v"(ldsXs), "v"(ldsXs + 16u), "v"(gp), "v"(gp + 4)
          : "memory");
    }
    // ---- 4x WMMA, K = 128
    v8f d = {};
#pragma unroll
    for (int ks = 0; ks < 4; ++ks) {
      d = __builtin_amdgcn_wmma_f32_16x16x32_f16(false, afrag[ks], false, bcur[ks],
                                                 (short)0, d, false, false);
    }
    __syncthreads();  // previous chunk's Ps/Xs consumers done
    // ---- scatter D (8 VGPRs: M = r / r+8 split at lane 16) into Zs
#pragma unroll
    for (int r = 0; r < 8; ++r) {
      const int ccr = mi * 16 + lhalf * 8 + r;
      const int ll  = ni * 16 + lmod;
      Zs[ccr * 32 + ll] = d[r];
    }
    asm volatile("s_wait_asynccnt 0x0" ::: "memory");  // Xs chunk landed
    __syncthreads();
    // ---- column softmax over c (64 rows) : 8 groups x 32 columns
    const int lc = t & 31;
    const int g  = t >> 5;
    float m = -3.4e38f;
#pragma unroll
    for (int i = 0; i < 8; ++i) m = fmaxf(m, Zs[(g * 8 + i) * 32 + lc]);
    pred[g * 32 + lc] = m;
    __syncthreads();
    if (g == 0) {
      float mm = pred[lc];
#pragma unroll
      for (int gg = 1; gg < 8; ++gg) mm = fmaxf(mm, pred[gg * 32 + lc]);
      colmax[lc] = mm;
    }
    __syncthreads();
    const float cm = colmax[lc];
    float s = 0.f;
#pragma unroll
    for (int i = 0; i < 8; ++i) s += __expf(Zs[(g * 8 + i) * 32 + lc] - cm);
    pred[g * 32 + lc] = s;
    __syncthreads();
    if (g == 0) {
      float ss = 0.f;
#pragma unroll
      for (int gg = 0; gg < 8; ++gg) ss += pred[gg * 32 + lc];
      colsum[lc] = ss;
    }
    __syncthreads();
    const float inv = 1.f / colsum[lc];
#pragma unroll
    for (int i = 0; i < 8; ++i) {
      const int idx = (g * 8 + i) * 32 + lc;
      Ps[idx] = __expf(Zs[idx] - cm) * inv;
    }
    __syncthreads();
    // ---- accumulate sum_l X[b,c,l] * P[c,l] (each thread: 8 l's of one c)
#pragma unroll
    for (int i = 0; i < 8; ++i) {
      const int ll = q * 8 + i;
      acc_out = fmaf(Xs[cacc * 32 + ll], Ps[cacc * 32 + ll], acc_out);
    }
    // rotate double buffer
#pragma unroll
    for (int ks = 0; ks < 4; ++ks) bcur[ks] = bnext[ks];
  }
  racc[cacc * 4 + q] = acc_out;
  __syncthreads();
  if (t < CC) {
    const float s = racc[t * 4] + racc[t * 4 + 1] + racc[t * 4 + 2] + racc[t * 4 + 3];
    Xout[((size_t)b * CC + t) * HID + h] = s;
  }
}

// --------------------------------------------------------------------------
// Kernel D: softmax-weighted pooling over C + T_bias + relu -> (B,1,HID)
// --------------------------------------------------------------------------
__global__ void __launch_bounds__(128) pool_kernel(const float* __restrict__ Xout,
                                                   const float* __restrict__ Tb,
                                                   float* __restrict__ out) {
  const int b  = blockIdx.x;
  const int hh = threadIdx.x;   // 0..127
  const float* p = Xout + (size_t)b * CC * HID + hh;
  float m = -3.4e38f;
  for (int c = 0; c < CC; ++c) m = fmaxf(m, p[c * HID]);
  float s = 0.f;
  for (int c = 0; c < CC; ++c) s += __expf(p[c * HID] - m);
  const float inv = 1.f / s;
  float pool = 0.f;
  for (int c = 0; c < CC; ++c) {
    const float v = p[c * HID];
    pool = fmaf(v, __expf(v - m) * inv, pool);
  }
  out[b * HID + hh] = fmaxf(pool + Tb[hh], 0.f);
}

// --------------------------------------------------------------------------
extern "C" void kernel_launch(void* const* d_in, const int* in_sizes, int n_in,
                              void* d_out, int out_size, void* d_ws, size_t ws_size,
                              hipStream_t stream) {
  const float* X  = (const float*)d_in[0];
  const float* W1 = (const float*)d_in[1];
  const float* b1 = (const float*)d_in[2];
  const float* W2 = (const float*)d_in[3];
  const float* b2 = (const float*)d_in[4];
  const float* W3 = (const float*)d_in[5];
  // d_in[6] = b3: constant across the channel-softmax axis -> cancels; unused.
  const float* Tb = (const float*)d_in[7];

  char* ws = (char*)d_ws;
  _Float16* W3t = (_Float16*)ws;                                  // 65536*128*2 = 16 MB
  _Float16* H2  = (_Float16*)(ws + (size_t)16777216);             // 2048*128*2 = 512 KB
  float*    Xo  = (float*)   (ws + (size_t)16777216 + 524288);    // 2048*128*4 = 1 MB

  hipLaunchKernelGGL(w3_convert_kernel, dim3(1024, 4), dim3(256), 0, stream, W3, W3t);
  hipLaunchKernelGGL(mlp_kernel, dim3(BB * CC), dim3(128), 0, stream, X, W1, b1, W2, b2, H2);
  hipLaunchKernelGGL(filt_apply_kernel, dim3(HID, BB), dim3(256), 0, stream, X, H2, W3t, Xo);
  hipLaunchKernelGGL(pool_kernel, dim3(BB), dim3(128), 0, stream, Xo, Tb, (float*)d_out);
}
